// RuleLayer_19387482374754
// MI455X (gfx1250) — matrix operand order repaired and checked by reading, not compiled, not run on hardware
//
#include <hip/hip_runtime.h>
#include <hip/hip_bf16.h>

typedef __attribute__((ext_vector_type(16))) _Float16 v16h;
typedef __attribute__((ext_vector_type(8)))  float    v8f;

// Problem sizes (fixed by reference)
#define BB   4096     // batch
#define RR   1024     // rules
#define FF   32       // features
#define MM   8        // memberships
#define KK   256      // F*M contraction dim
#define BT   256      // batch tiles of 16
#define NT   64       // rule  tiles of 16
#define KB   8        // k-blocks of 32

// ---------------------------------------------------------------------------
// Build swizzled one-hot selection matrix S[k][r] in f16, laid out so each
// lane loads its B-fragment (v16h) as one contiguous 32B chunk.
// B-layout (16-bit B 32x16, wave32): lane L -> n = L%16, g = L/16;
// half j -> k = kb*32 + (j<8?0:16) + 8*g + (j&7).
// ---------------------------------------------------------------------------
__global__ __launch_bounds__(256) void prep_S(const int* __restrict__ idx,
                                              _Float16* __restrict__ Smat) {
    int flat = blockIdx.x * 256 + threadIdx.x;   // 0..16383
    int lane = flat & 31;
    int grp  = flat >> 5;                        // (ntile*8 + kb), 0..511
    int kb    = grp & 7;
    int ntile = grp >> 3;
    int n = lane & 15, g = lane >> 4;
    int r = ntile * 16 + n;
    int kbase = kb * 32 + g * 8;
    v16h s;
#pragma unroll
    for (int j = 0; j < 16; ++j) {
        int k = kbase + ((j < 8) ? 0 : 16) + (j & 7);
        int f = k >> 3, m = k & 7;
        s[j] = (idx[r * FF + f] == m) ? (_Float16)1.0f : (_Float16)0.0f;
    }
    *(v16h*)(Smat + ((size_t)grp * 32 + lane) * 16) = s;
}

// ---------------------------------------------------------------------------
// Compute L = log(mf + 1e-9) split into f16 hi + f16 residual lo, written in
// swizzled A-fragment layout.
// A-layout (16-bit A 16x32, wave32): lane L -> m = L%16, g = L/16;
// half j -> k = kb*32 + (j<8?0:16) + 8*g + (j&7).
// ---------------------------------------------------------------------------
__global__ __launch_bounds__(256) void prep_A(const float* __restrict__ mf,
                                              _Float16* __restrict__ Ahi,
                                              _Float16* __restrict__ Alo) {
    int flat = blockIdx.x * 256 + threadIdx.x;   // 0..65535
    int lane = flat & 31;
    int grp  = flat >> 5;                        // (btile*8 + kb), 0..2047
    int kb    = grp & 7;
    int btile = grp >> 3;
    int mrow = lane & 15, g = lane >> 4;
    const float* src = mf + (size_t)(btile * 16 + mrow) * KK + kb * 32 + 8 * g;
    v16h hi, lo;
#pragma unroll
    for (int j = 0; j < 16; ++j) {
        int k_off = ((j < 8) ? 0 : 16) + (j & 7);
        float x = logf(src[k_off] + 1e-9f);
        _Float16 h = (_Float16)x;
        hi[j] = h;
        lo[j] = (_Float16)(x - (float)h);
    }
    size_t o = ((size_t)grp * 32 + lane) * 16;
    *(v16h*)(Ahi + o) = hi;
    *(v16h*)(Alo + o) = lo;
}

// ---------------------------------------------------------------------------
// GEMM + exp + row-normalization.
// Block = 256 threads (8 waves). Block owns 32 batch rows x all 1024 rules.
// wave&1 -> which 16-row btile; wave>>1 -> which quarter of the 64 ntiles.
// Per 16x16 tile: 8 k-blocks x 2 (hi residual + lo residual) WMMAs.
// ---------------------------------------------------------------------------
__global__ __launch_bounds__(256) void fuzzy_gemm(const _Float16* __restrict__ Ahi,
                                                  const _Float16* __restrict__ Alo,
                                                  const _Float16* __restrict__ Smat,
                                                  float* __restrict__ out) {
    __shared__ float rowsum[32];
    const int tid  = threadIdx.x;
    const int wave = tid >> 5;
    const int lane = tid & 31;
    if (tid < 32) rowsum[tid] = 0.0f;
    __syncthreads();

    const int btileL = wave & 1;                 // local btile 0/1
    const int btile  = blockIdx.x * 2 + btileL;  // global btile
    const int waveN  = wave >> 1;                // 0..3

    // Register-resident A fragments for this wave's 16 batch rows.
    v16h ah[KB], al[KB];
    {
        const _Float16* abh = Ahi + ((size_t)btile * KB) * 512 + lane * 16;
        const _Float16* abl = Alo + ((size_t)btile * KB) * 512 + lane * 16;
#pragma unroll
        for (int kb = 0; kb < KB; ++kb) {
            ah[kb] = *(const v16h*)(abh + kb * 512);
            al[kb] = *(const v16h*)(abl + kb * 512);
        }
    }

    float psum[8];
#pragma unroll
    for (int v = 0; v < 8; ++v) psum[v] = 0.0f;

    const int g = lane >> 4;
    const int nlane = lane & 15;

    for (int i = 0; i < 16; ++i) {
        const int ntile = waveN + 4 * i;         // 0..63
        const _Float16* sbase = Smat + (size_t)ntile * 4096 + lane * 16;
        v8f c = {};
#pragma unroll
        for (int kb = 0; kb < KB; ++kb) {
            v16h s = *(const v16h*)(sbase + kb * 512);
            // c += A_hi * S ; c += A_lo * S   (f32 accumulate)
            c = __builtin_amdgcn_wmma_f32_16x16x32_f16(
                    false, ah[kb], false, s, (short)0, c, false, false);
            c = __builtin_amdgcn_wmma_f32_16x16x32_f16(
                    false, al[kb], false, s, (short)0, c, false, false);
        }
        // firing = exp(log_firing); store + accumulate row partial sums.
        const int n = ntile * 16 + nlane;
#pragma unroll
        for (int v = 0; v < 8; ++v) {
            float e = __expf(c[v]);
            psum[v] += e;
            int m = v + 8 * g;                   // C layout: row = v + 8*(lane/16)
            out[(size_t)(btile * 16 + m) * RR + n] = e;
        }
    }

    // Reduce over the 16 lanes that share a row half (xor 1,2,4,8 stays in half).
#pragma unroll
    for (int v = 0; v < 8; ++v) {
        float val = psum[v];
#pragma unroll
        for (int off = 1; off < 16; off <<= 1)
            val += __shfl_xor(val, off, 32);
        if (nlane == 0)
            atomicAdd(&rowsum[btileL * 16 + v + 8 * g], val);  // ds_add_f32
    }

    __threadfence();
    __syncthreads();

    // Normalize: iteration i handles the whole local row i (1024 floats) with
    // 256 lanes x float4 -> fully coalesced.
    float* nrm = out + (size_t)BB * RR;
    const float4* of4 = (const float4*)(out + (size_t)blockIdx.x * 32 * RR);
    float4*       nf4 = (float4*)(nrm + (size_t)blockIdx.x * 32 * RR);
#pragma unroll 4
    for (int i = 0; i < 32; ++i) {
        float inv = 1.0f / (rowsum[i] + 1e-6f);
        float4 f = of4[i * 256 + tid];
        nf4[i * 256 + tid] = make_float4(f.x * inv, f.y * inv, f.z * inv, f.w * inv);
    }
}

// ---------------------------------------------------------------------------
extern "C" void kernel_launch(void* const* d_in, const int* in_sizes, int n_in,
                              void* d_out, int out_size, void* d_ws, size_t ws_size,
                              hipStream_t stream) {
    const float* mf  = (const float*)d_in[0];   // (B, F, M) f32
    const int*   idx = (const int*)d_in[1];     // (R, F) i32
    float*       out = (float*)d_out;           // firing (B*R) then norm (B*R)

    // Workspace: A_hi (2MB) | A_lo (2MB) | S (0.5MB)  in f16
    _Float16* Ahi  = (_Float16*)d_ws;
    _Float16* Alo  = Ahi + (size_t)BT * KB * 512;   // 1,048,576 halfs
    _Float16* Smat = Alo + (size_t)BT * KB * 512;   //   262,144 halfs

    prep_S<<<64, 256, 0, stream>>>(idx, Smat);
    prep_A<<<256, 256, 0, stream>>>(mf, Ahi, Alo);
    fuzzy_gemm<<<128, 256, 0, stream>>>(Ahi, Alo, Smat, out);
}
